// DiscreteGaussianHMM_76862734730019
// MI455X (gfx1250) — compile-verified, matrix operand order checked by reading, not run on hardware
//
#include <hip/hip_runtime.h>
#include <math.h>

#define T_DIM 512
#define B_DIM 256
#define S_DIM 256
#define A_DIM 8
#define D_DIM 64
#define KMIX  (A_DIM * S_DIM)   /* 2048 */
#define EPSV  1e-6f
#define LOG2PI 1.8378770664093453f

typedef __attribute__((ext_vector_type(16))) __bf16 v16bf;
typedef __attribute__((ext_vector_type(8)))  float  v8f;
typedef __attribute__((ext_vector_type(2)))  float  v2f;

// ---------------- block reductions (blockDim.x == 256) ----------------
__device__ __forceinline__ float block_reduce_max(float v, float* sb) {
  int t = threadIdx.x;
  sb[t] = v; __syncthreads();
  #pragma unroll
  for (int off = 128; off > 0; off >>= 1) {
    if (t < off) sb[t] = fmaxf(sb[t], sb[t + off]);
    __syncthreads();
  }
  float r = sb[0]; __syncthreads();
  return r;
}
__device__ __forceinline__ float block_reduce_sum(float v, float* sb) {
  int t = threadIdx.x;
  sb[t] = v; __syncthreads();
  #pragma unroll
  for (int off = 128; off > 0; off >>= 1) {
    if (t < off) sb[t] += sb[t + off];
    __syncthreads();
  }
  float r = sb[0]; __syncthreads();
  return r;
}

// ---------------- setup: softmax(w_logits) -> packed bf16 B-fragments ----------------
// Forward buffer Wf: K=(a*256+i), N=j, value trans[a,i,j]
// Backward buffer Wb: K=(a*256+j), N=i, value trans[a,i,j]+EPS  (transposed per a)
// Pack layout for V_WMMA_*_16x16x32 B operand (32x16 tile):
//   tile (kt,nt); lane = n_local + 16*((k_local>>3)&1); elem j: k = 16*(j>>3)+8*half+(j&7)
__device__ __forceinline__ void pack_b16(__bf16* W, int k, int n, float val) {
  int kt = k >> 5, kl = k & 31, nt = n >> 4, nl = n & 15;
  int lane = nl + (((kl >> 3) & 1) << 4);
  int jj   = (((kl >> 4) & 1) << 3) | (kl & 7);
  W[(size_t)((kt * (S_DIM / 16) + nt) * 32 + lane) * 16 + jj] = (__bf16)val;
}

__global__ void k_prep_trans(const float* __restrict__ wlog,
                             __bf16* __restrict__ Wf, __bf16* __restrict__ Wb) {
  __shared__ float sb[256];
  int r = blockIdx.x;              // (a,i) row in [0, A*S)
  int a = r >> 8, i = r & 255;
  int j = threadIdx.x;
  float v = wlog[(size_t)r * S_DIM + j];
  float m = block_reduce_max(v, sb);
  float e = __expf(v - m);
  float s = block_reduce_sum(e, sb);
  float p = e / s;
  pack_b16(Wf, (a << 8) | i, j, p);          // forward: row-stochastic trans
  pack_b16(Wb, (a << 8) | j, i, p + EPSV);   // backward: trans^T + EPS
}

// ---------------- setup: emission coefficient matrix (f32, 4x16 B-frag packing) ----------------
__global__ void k_prep_emis(const float* __restrict__ mu, const float* __restrict__ lv,
                            float* __restrict__ Wlp, float* __restrict__ bias) {
  int s = threadIdx.x;
  float quadc = 0.f, suml = 0.f;
  for (int d = 0; d < D_DIM; ++d) {
    float l  = lv[(size_t)s * D_DIM + d];
    float iv = __expf(-l);
    float m  = mu[(size_t)s * D_DIM + d];
    quadc += m * m * iv;
    suml  += l;
    float coef[2] = { m * iv, -0.5f * iv };
    int   kk[2]   = { d, d + D_DIM };
    #pragma unroll
    for (int q = 0; q < 2; ++q) {
      int k = kk[q], n = s;
      int kt = k >> 2, kl = k & 3, nt = n >> 4, nl = n & 15;
      int lane = nl + ((kl >> 1) << 4);
      int jj   = kl & 1;
      Wlp[((kt * (S_DIM / 16) + nt) * 32 + lane) * 2 + jj] = coef[q];
    }
  }
  bias[s] = -0.5f * (quadc + suml + (float)D_DIM * LOG2PI);
}

__global__ void k_prep_pi(const float* __restrict__ s0, float* __restrict__ logpi0) {
  __shared__ float sb[256];
  int j = threadIdx.x;
  float v = s0[j];
  float m = block_reduce_max(v, sb);
  float e = __expf(v - m);
  float s = block_reduce_sum(e, sb);
  logpi0[j] = v - m - __logf(s);
}

// ---------------- emission GEMM: [TB,128] x [128,S] via V_WMMA_F32_16X16X4_F32 ----------------
__global__ void k_emis_gemm(const float* __restrict__ x, const float* __restrict__ Wlp,
                            const float* __restrict__ bias, float* __restrict__ logp) {
  int lane = threadIdx.x & 31;
  int wid  = threadIdx.x >> 5;
  int rowbase = (blockIdx.x * 4 + wid) << 4;
  int nt = blockIdx.y, colbase = nt << 4;
  const float* xrow = x + (size_t)(rowbase + (lane & 15)) * D_DIM;
  int kofs = (lane >> 4) << 1;   // A 16x4 f32 frag: lanes 0-15 hold K={0,1}, 16-31 K={2,3}
  v8f acc = {};
  #pragma unroll
  for (int kt = 0; kt < 16; ++kt) {            // K 0..63 : x
    v2f av = *(const v2f*)(xrow + kt * 4 + kofs);
    v2f bv = *(const v2f*)(Wlp + ((kt * 16 + nt) * 32 + lane) * 2);
    acc = __builtin_amdgcn_wmma_f32_16x16x4_f32(false, av, false, bv,
                                                (short)0, acc, false, false);
  }
  #pragma unroll
  for (int kt = 16; kt < 32; ++kt) {           // K 64..127 : x^2
    v2f t  = *(const v2f*)(xrow + (kt - 16) * 4 + kofs);
    v2f av = t * t;
    v2f bv = *(const v2f*)(Wlp + ((kt * 16 + nt) * 32 + lane) * 2);
    acc = __builtin_amdgcn_wmma_f32_16x16x4_f32(false, av, false, bv,
                                                (short)0, acc, false, false);
  }
  float bcol = bias[colbase + (lane & 15)];
  #pragma unroll
  for (int r = 0; r < 8; ++r) {
    int row = rowbase + r + ((lane >> 4) << 3);
    logp[(size_t)row * S_DIM + colbase + (lane & 15)] = acc[r] + bcol;
  }
}

// ---------------- fused mixture GEMM ----------------
// Computes vout[b,n] = sum_{a,k} (aw[b,a]*base_row[b,k]) * Bpk[(a,k),n]
// A' fragments are built inline: 8 contiguous f32 from base_row scaled by the
// per-row mixture weight (segment index is compile-time under full unroll),
// converted to bf16, fed to V_WMMA_F32_16X16X32_BF16.
__global__ void k_mix_gemm(const float* __restrict__ rowsrc, const float* __restrict__ a_prev,
                           const __bf16* __restrict__ Bpk, float* __restrict__ vout) {
  int lane = threadIdx.x & 31;
  int wid  = threadIdx.x >> 5;
  int mt = blockIdx.x * 4 + wid;
  int nt = blockIdx.y;
  int khalf = lane >> 4;
  int r_ = (mt << 4) + (lane & 15);
  const float* rowptr = rowsrc + (size_t)r_ * S_DIM;
  float aw[A_DIM];
  #pragma unroll
  for (int ai = 0; ai < A_DIM; ++ai) aw[ai] = a_prev[r_ * A_DIM + ai];
  v8f acc = {};
  #pragma unroll
  for (int kt = 0; kt < KMIX / 32; ++kt) {
    const int base1 = kt * 32;            // + 8*khalf at runtime
    const int base2 = kt * 32 + 16;
    const float w1 = aw[base1 >> 8];      // compile-time segment index
    const float w2 = aw[base2 >> 8];      // (8-elem run never crosses a 256 boundary)
    __builtin_prefetch(rowptr + ((base1 + 256) & 255), 0, 1);  // global_prefetch_b8
    v8f lo = *(const v8f*)(rowptr + (base1 & 255) + (khalf << 3));
    v8f hi = *(const v8f*)(rowptr + (base2 & 255) + (khalf << 3));
    v16bf af;
    #pragma unroll
    for (int e = 0; e < 8; ++e) {
      af[e]     = (__bf16)(w1 * lo[e]);
      af[e + 8] = (__bf16)(w2 * hi[e]);
    }
    v16bf bfr = *(const v16bf*)(Bpk + (size_t)((kt * 16 + nt) * 32 + lane) * 16);
    acc = __builtin_amdgcn_wmma_f32_16x16x32_bf16(false, af, false, bfr,
                                                  (short)0, acc, false, false);
  }
  #pragma unroll
  for (int r = 0; r < 8; ++r) {
    int orow = (mt << 4) + r + (khalf << 3);
    vout[(size_t)orow * S_DIM + (nt << 4) + (lane & 15)] = acc[r];
  }
}

// ---------------- forward pieces ----------------
__global__ void k_init_alpha(const float* __restrict__ logp0, const float* __restrict__ logpi0,
                             float* __restrict__ alpha0) {
  __shared__ float sb[256];
  int b = blockIdx.x, j = threadIdx.x;
  float v = logp0[(size_t)b * S_DIM + j] + logpi0[j];
  float m = block_reduce_max(v, sb);
  float e = __expf(v - m);
  float s = block_reduce_sum(e, sb);
  alpha0[(size_t)b * S_DIM + j] = e / s;
}

__global__ void k_fwd_epilogue(const float* __restrict__ vbuf, const float* __restrict__ logp_t,
                               float* __restrict__ alpha_t) {
  __shared__ float sb[256];
  int b = blockIdx.x, j = threadIdx.x;
  float v = __logf(vbuf[(size_t)b * S_DIM + j] + (float)S_DIM * EPSV)
          + logp_t[(size_t)b * S_DIM + j];
  float m = block_reduce_max(v, sb);
  float e = __expf(v - m);
  float s = block_reduce_sum(e, sb);
  alpha_t[(size_t)b * S_DIM + j] = e / s;
}

// ---------------- backward pieces ----------------
// Start: q_z[T-1] = softmax(log(alpha+EPS)); seed w = exp(logp[T-1] - M) (logb == 0).
__global__ void k_bwd_start(const float* __restrict__ alpha_last, const float* __restrict__ logp_last,
                            float* __restrict__ qz_last, float* __restrict__ wbuf,
                            float* __restrict__ Mbuf) {
  __shared__ float sb[256];
  int b = blockIdx.x, j = threadIdx.x;
  float v = __logf(alpha_last[(size_t)b * S_DIM + j] + EPSV);
  float m = block_reduce_max(v, sb);
  float e = __expf(v - m);
  float s = block_reduce_sum(e, sb);
  qz_last[(size_t)b * S_DIM + j] = e / s;
  float c  = logp_last[(size_t)b * S_DIM + j];
  float cm = block_reduce_max(c, sb);
  wbuf[(size_t)b * S_DIM + j] = __expf(c - cm);
  if (j == 0) Mbuf[b] = cm;
}

// Step (for t): logb_{t-1} = (log(u)+M)*mask; q_z[t-1] = softmax(log(alpha[t-1]+EPS)+logb);
// then prepare next step's w = exp(logp[t-1]+logb - M') in-place (logb never stored).
__global__ void k_bwd_step(const float* __restrict__ u, const float* __restrict__ mask_t,
                           const float* __restrict__ alpha_prev, const float* __restrict__ logp_prev,
                           float* __restrict__ qz_prev, float* __restrict__ wbuf,
                           float* __restrict__ Mbuf) {
  __shared__ float sb[256];
  int b = blockIdx.x, j = threadIdx.x;
  float lb = __logf(u[(size_t)b * S_DIM + j]) + Mbuf[b];
  lb = (mask_t[b] == 1.0f) ? lb : 0.0f;
  // posterior at t-1
  float v = __logf(alpha_prev[(size_t)b * S_DIM + j] + EPSV) + lb;
  float m = block_reduce_max(v, sb);
  float e = __expf(v - m);
  float s = block_reduce_sum(e, sb);
  qz_prev[(size_t)b * S_DIM + j] = e / s;
  // fused next-step build: c = logp[t-1] + logb_{t-1}
  float c  = logp_prev[(size_t)b * S_DIM + j] + lb;
  float cm = block_reduce_max(c, sb);
  wbuf[(size_t)b * S_DIM + j] = __expf(c - cm);
  if (j == 0) Mbuf[b] = cm;
}

// ---------------- host orchestration ----------------
extern "C" void kernel_launch(void* const* d_in, const int* in_sizes, int n_in,
                              void* d_out, int out_size, void* d_ws, size_t ws_size,
                              hipStream_t stream) {
  (void)in_sizes; (void)n_in; (void)out_size; (void)ws_size;
  const float* x    = (const float*)d_in[0];
  const float* a    = (const float*)d_in[1];
  const float* mask = (const float*)d_in[2];
  const float* mu   = (const float*)d_in[3];
  const float* lv   = (const float*)d_in[4];
  const float* wlog = (const float*)d_in[5];
  const float* s0   = (const float*)d_in[6];
  float* qz = (float*)d_out;

  char* p = (char*)d_ws;
  auto carve = [&](size_t bytes) -> char* {
    char* r = p; p += (bytes + 255) & ~(size_t)255; return r;
  };
  float*  logp   = (float*)carve((size_t)T_DIM * B_DIM * S_DIM * 4);
  float*  alpha  = (float*)carve((size_t)T_DIM * B_DIM * S_DIM * 4);
  float*  wbuf   = (float*)carve((size_t)B_DIM * S_DIM * 4);
  float*  vbuf   = (float*)carve((size_t)B_DIM * S_DIM * 4);
  float*  Mbuf   = (float*)carve((size_t)B_DIM * 4);
  float*  bias   = (float*)carve((size_t)S_DIM * 4);
  float*  logpi0 = (float*)carve((size_t)S_DIM * 4);
  float*  Wlp    = (float*)carve((size_t)128 * S_DIM * 4);
  __bf16* Wf     = (__bf16*)carve((size_t)KMIX * S_DIM * 2);
  __bf16* Wb     = (__bf16*)carve((size_t)KMIX * S_DIM * 2);

  // one-time prep
  k_prep_trans<<<A_DIM * S_DIM, 256, 0, stream>>>(wlog, Wf, Wb);
  k_prep_emis<<<1, S_DIM, 0, stream>>>(mu, lv, Wlp, bias);
  k_prep_pi<<<1, S_DIM, 0, stream>>>(s0, logpi0);

  // emissions: one f32-WMMA GEMM over all T*B rows
  k_emis_gemm<<<dim3((T_DIM * B_DIM) / 64, S_DIM / 16), 128, 0, stream>>>(x, Wlp, bias, logp);

  // forward scan (2 kernels per step; A' built inside the GEMM)
  k_init_alpha<<<B_DIM, 256, 0, stream>>>(logp, logpi0, alpha);
  for (int t = 1; t < T_DIM; ++t) {
    k_mix_gemm<<<dim3(4, 16), 128, 0, stream>>>(
        alpha + (size_t)(t - 1) * B_DIM * S_DIM,
        a + (size_t)(t - 1) * B_DIM * A_DIM, Wf, vbuf);
    k_fwd_epilogue<<<B_DIM, 256, 0, stream>>>(
        vbuf, logp + (size_t)t * B_DIM * S_DIM,
        alpha + (size_t)t * B_DIM * S_DIM);
  }

  // backward scan + fused posterior (2 kernels per step; no logb / A' buffers)
  k_bwd_start<<<B_DIM, 256, 0, stream>>>(
      alpha + (size_t)(T_DIM - 1) * B_DIM * S_DIM,
      logp + (size_t)(T_DIM - 1) * B_DIM * S_DIM,
      qz + (size_t)(T_DIM - 1) * B_DIM * S_DIM, wbuf, Mbuf);
  for (int t = T_DIM - 1; t >= 1; --t) {
    k_mix_gemm<<<dim3(4, 16), 128, 0, stream>>>(
        wbuf, a + (size_t)(t - 1) * B_DIM * A_DIM, Wb, vbuf);
    k_bwd_step<<<B_DIM, 256, 0, stream>>>(
        vbuf, mask + (size_t)t * B_DIM,
        alpha + (size_t)(t - 1) * B_DIM * S_DIM,
        logp + (size_t)(t - 1) * B_DIM * S_DIM,
        qz + (size_t)(t - 1) * B_DIM * S_DIM, wbuf, Mbuf);
  }
}
